// toy_fasterrcnn_52639119180276
// MI455X (gfx1250) — matrix operand order repaired and checked by reading, not compiled, not run on hardware
//
#include <hip/hip_runtime.h>

typedef __attribute__((ext_vector_type(16))) _Float16 v16h;
typedef __attribute__((ext_vector_type(8)))  _Float16 v8h;
typedef __attribute__((ext_vector_type(8)))  float    v8f;

#define BB     4
#define CIN    512
#define IHH    96
#define IWW    160
#define PH     98            // padded height
#define PW     162           // padded width
#define COUT   512
#define OH     48
#define OW     80
#define HWo    (OH*OW)        // 3840
#define KTOT   4608           // 512*9
#define NSTEPS 144            // KTOT/32
#define NANCH  9
#define NPROP  (HWo*NANCH)    // 34560
#define PRE_K  2000
#define POST_K 300
#define NEGV   (-1e30f)

// anchors from _generate_anchors(16,(0.5,1,2),(8,16,32)) (numpy round-half-even)
__constant__ float ANCH[9][4] = {
  { -84.f,  -40.f,  99.f,  55.f}, {-176.f,  -88.f, 191.f, 103.f}, {-360.f, -184.f, 375.f, 199.f},
  { -56.f,  -56.f,  71.f,  71.f}, {-120.f, -120.f, 135.f, 135.f}, {-248.f, -248.f, 263.f, 263.f},
  { -36.f,  -80.f,  51.f,  95.f}, { -80.f, -168.f,  95.f, 183.f}, {-168.f, -344.f, 183.f, 359.f}
};

__device__ inline unsigned ordf(float f) {
  unsigned u = __float_as_uint(f);
  return (u & 0x80000000u) ? ~u : (u | 0x80000000u);
}

// ---------------- layout conversion ----------------
// x (NCHW f32) -> zero-padded NHWC f16: xp[b][ihp][iwp][ci], ihp in [0,98), iwp in [0,162)
__global__ __launch_bounds__(256)
void convert_x_k(const float* __restrict__ src, _Float16* __restrict__ dst) {
  int i = blockIdx.x * 256 + threadIdx.x;
  if (i >= BB * PH * PW * CIN) return;
  int ci  = i % CIN;
  int t   = i / CIN;
  int iwp = t % PW;
  int t2  = t / PW;
  int ihp = t2 % PH;
  int b   = t2 / PH;
  float v = 0.f;
  if (ihp >= 1 && ihp <= IHH && iwp >= 1 && iwp <= IWW)
    v = src[(((size_t)b * CIN + ci) * IHH + (ihp - 1)) * IWW + (iwp - 1)];
  dst[i] = (_Float16)v;
}

__global__ __launch_bounds__(256)
void convert_w_k(const float* __restrict__ src, _Float16* __restrict__ dst) {
  int i = blockIdx.x * 256 + threadIdx.x;
  if (i >= COUT * KTOT) return;                   // dst[co][kykx*512+ci]
  int k    = i % KTOT;
  int co   = i / KTOT;
  int kykx = k / CIN;
  int ci   = k % CIN;
  dst[i] = (_Float16)src[((size_t)co * CIN + ci) * 9 + kykx];
}

// ---------------- conv1: 3x3 s2 512->512, implicit-GEMM WMMA f16, double-buffered ----------------
// Block: 64 M x 256 N, 8 waves; wave: 2 m-tiles x 4 n-tiles (8 accumulators).
__global__ __launch_bounds__(256)
void conv1_wmma(const _Float16* __restrict__ xp, const _Float16* __restrict__ w1t,
                const float* __restrict__ b1, float* __restrict__ feat) {
  const int lane = threadIdx.x & 31;
  const int wave = threadIdx.x >> 5;
  const int row  = lane & 15;
  const int hi   = lane >> 4;

  const int mblk  = blockIdx.x;       // 0..239
  const int nblk  = blockIdx.y;       // 0..1
  const int mpair = wave >> 2;        // 0..1
  const int nquad = wave & 3;         // 0..3
  const int nbase = nblk * 256 + nquad * 64;

  int mb[2], moh[2], mow[2];
  const _Float16* baseA[2];
#pragma unroll
  for (int mi = 0; mi < 2; ++mi) {
    const int m0  = mblk * 64 + mpair * 32 + mi * 16;   // 16-aligned, OW%16==0 -> single row
    const int b   = m0 / HWo;
    const int rem = m0 % HWo;
    mb[mi]  = b;
    moh[mi] = rem / OW;
    mow[mi] = rem % OW;
    const int ow_lane = mow[mi] + row;
    // padded coords: ihp = oh*2 (+ky via aoff), iwp = ow_lane*2 (+kx via aoff)
    baseA[mi] = xp + ((size_t)((b * PH + moh[mi] * 2) * PW + ow_lane * 2) * CIN) + hi * 8;
  }
  const _Float16* baseB[4];
#pragma unroll
  for (int j = 0; j < 4; ++j)
    baseB[j] = w1t + (size_t)(nbase + j * 16 + row) * KTOT + hi * 8;

  union AU { v16h v; v8h h[2]; };
  AU a0[2], bm0[4], a1[2], bm1[4];
  v8f acc[2][4] = {};

  // flat K-step ks in [0,144): B offset = ks*32; A offset = ks*32 + (ks/48)*81408
  // (81408 = (PW-3)*CIN covers the ky row jump)
  auto loadstep = [&](int ks, AU (&aa)[2], AU (&bb)[4]) {
    const int aoff = ks * 32 + (ks / 48) * ((PW - 3) * CIN);
#pragma unroll
    for (int mi = 0; mi < 2; ++mi) {
      // A 16x32 f16 fragment: element e = row[K = (e/8)*16 + hi*8 + e%8]
      aa[mi].h[0] = *(const v8h*)(baseA[mi] + aoff);
      aa[mi].h[1] = *(const v8h*)(baseA[mi] + aoff + 16);
    }
    const int boff = ks * 32;
#pragma unroll
    for (int j = 0; j < 4; ++j) {
      // B 32x16 f16 fragment: lane holds column n, element e = col[K = (e/8)*16 + hi*8 + e%8]
      bb[j].h[0] = *(const v8h*)(baseB[j] + boff);
      bb[j].h[1] = *(const v8h*)(baseB[j] + boff + 16);
    }
  };

  auto mmastep = [&](AU (&aa)[2], AU (&bb)[4]) {
#pragma unroll
    for (int mi = 0; mi < 2; ++mi)
#pragma unroll
      for (int j = 0; j < 4; ++j)
        acc[mi][j] = __builtin_amdgcn_wmma_f32_16x16x32_f16(
            false, aa[mi].v, false, bb[j].v, (short)0, acc[mi][j], false, false);
  };

  // software pipeline, manually unrolled x2 so buffer indices are compile-time
  loadstep(0, a0, bm0);
  for (int ks = 0; ks < NSTEPS - 2; ks += 2) {
    loadstep(ks + 1, a1, bm1);
    mmastep(a0, bm0);
    loadstep(ks + 2, a0, bm0);
    mmastep(a1, bm1);
  }
  loadstep(NSTEPS - 1, a1, bm1);
  mmastep(a0, bm0);
  mmastep(a1, bm1);

  const int mh = hi * 8;   // D: (M = r + 8*hi, N = lane%16)
#pragma unroll
  for (int mi = 0; mi < 2; ++mi) {
#pragma unroll
    for (int j = 0; j < 4; ++j) {
      const int n = nbase + j * 16 + row;
      const float bias = b1[n];
      float* op = feat + (((size_t)(mb[mi] * COUT + n) * OH + moh[mi]) * OW + mow[mi] + mh);
#pragma unroll
      for (int r = 0; r < 8; ++r) op[r] = acc[mi][j][r] + bias;
    }
  }
}

// ---------------- 1x1 conv 512->36 (deltas) ----------------
__global__ __launch_bounds__(256)
void conv_1x1_k(const float* __restrict__ feat, const float* __restrict__ w,
                const float* __restrict__ bias, float* __restrict__ out) {
  int i = blockIdx.x * 256 + threadIdx.x;
  if (i >= BB * 36 * HWo) return;
  int hw = i % HWo;
  int t  = i / HWo;
  int co = t % 36;
  int b  = t / 36;
  const float* fp = feat + (size_t)b * COUT * HWo + hw;
  const float* wp = w + (size_t)co * COUT;
  float s = bias[co];
  for (int ci = 0; ci < COUT; ++ci) s += fp[(size_t)ci * HWo] * wp[ci];
  out[i] = s;
}

// ---------------- 3x3 pad1 conv 512->18 (cls) ----------------
__global__ __launch_bounds__(256)
void conv_cls_k(const float* __restrict__ feat, const float* __restrict__ w,
                const float* __restrict__ bias, float* __restrict__ out) {
  int i = blockIdx.x * 256 + threadIdx.x;
  if (i >= BB * 18 * HWo) return;
  int hw = i % HWo;
  int t  = i / HWo;
  int co = t % 18;
  int b  = t / 18;
  int oh = hw / OW, ow = hw % OW;
  float s = bias[co];
  for (int ky = 0; ky < 3; ++ky) {
    int ih = oh - 1 + ky;
    if (ih < 0 || ih >= OH) continue;
    for (int kx = 0; kx < 3; ++kx) {
      int iw = ow - 1 + kx;
      if (iw < 0 || iw >= OW) continue;
      const float* fp = feat + ((size_t)(b * COUT) * OH + ih) * OW + iw;
      const float* wp = w + (size_t)(co * COUT) * 9 + ky * 3 + kx;
      for (int ci = 0; ci < COUT; ++ci)
        s += fp[(size_t)ci * HWo] * wp[(size_t)ci * 9];
    }
  }
  out[i] = s;
}

// ---------------- anchor decode + clip + min-size filter ----------------
__global__ __launch_bounds__(256)
void decode_k(const float* __restrict__ deltas, const float* __restrict__ cls,
              float* __restrict__ boxes, float* __restrict__ scores) {
  int i = blockIdx.x * 256 + threadIdx.x;
  if (i >= BB * NPROP) return;
  int a = i % NANCH;
  int p = (i / NANCH) % HWo;
  int b = i / NPROP;
  int oh = p / OW, ow = p % OW;
  float gx = ow * 32.f, gy = oh * 32.f;
  float ax1 = gx + ANCH[a][0], ay1 = gy + ANCH[a][1];
  float ax2 = gx + ANCH[a][2], ay2 = gy + ANCH[a][3];
  float wsz = ax2 - ax1 + 1.f, hsz = ay2 - ay1 + 1.f;
  float cx = ax1 + 0.5f * wsz, cy = ay1 + 0.5f * hsz;
  size_t dbase = (((size_t)b * 36 + a * 4) * OH + oh) * OW + ow;
  float d0 = deltas[dbase];
  float d1 = deltas[dbase + (size_t)HWo];
  float d2 = deltas[dbase + (size_t)2 * HWo];
  float d3 = deltas[dbase + (size_t)3 * HWo];
  float pcx = d0 * wsz + cx, pcy = d1 * hsz + cy;
  float pw = expf(d2) * wsz, ph = expf(d3) * hsz;
  float x1 = pcx - 0.5f * pw, y1 = pcy - 0.5f * ph;
  float x2 = pcx + 0.5f * pw, y2 = pcy + 0.5f * ph;
  const float limx = 2559.f, limy = 1535.f;   // im = (1536, 2560)
  x1 = fminf(fmaxf(x1, 0.f), limx); x2 = fminf(fmaxf(x2, 0.f), limx);
  y1 = fminf(fmaxf(y1, 0.f), limy); y2 = fminf(fmaxf(y2, 0.f), limy);
  float bw = x2 - x1 + 1.f, bh = y2 - y1 + 1.f;
  float sc = cls[(((size_t)b * 18 + 9 + a) * OH + oh) * OW + ow];
  if (!(bw >= 16.f && bh >= 16.f)) sc = NEGV;
  size_t o = (size_t)b * NPROP + (size_t)p * NANCH + a;
  float* bo = boxes + o * 4;
  bo[0] = x1; bo[1] = y1; bo[2] = x2; bo[3] = y2;
  scores[o] = sc;
}

// ---------------- per-batch top-2000 threshold (bitwise binary search) ----------------
__global__ __launch_bounds__(256)
void thresh_k(const float* __restrict__ scores, int* __restrict__ tg) {
  int b = blockIdx.x, tid = threadIdx.x;
  const float* s = scores + (size_t)b * NPROP;
  __shared__ int cnt;
  unsigned T = 0;
  for (int bit = 31; bit >= 0; --bit) {
    unsigned c = T | (1u << bit);
    if (tid == 0) cnt = 0;
    __syncthreads();
    int local = 0;
    for (int i = tid; i < NPROP; i += 256) local += (ordf(s[i]) >= c) ? 1 : 0;
    atomicAdd(&cnt, local);
    __syncthreads();
    if (cnt >= PRE_K) T = c;
    __syncthreads();
  }
  if (tid == 0) cnt = 0;
  __syncthreads();
  int local = 0;
  for (int i = tid; i < NPROP; i += 256) local += (ordf(s[i]) > T) ? 1 : 0;
  atomicAdd(&cnt, local);
  __syncthreads();
  if (tid == 0) { tg[b * 2] = (int)T; tg[b * 2 + 1] = cnt; }
}

// ---------------- compact candidate indices ----------------
__global__ __launch_bounds__(256)
void compact_k(const float* __restrict__ scores, const int* __restrict__ tg,
               int* __restrict__ cand) {
  int b = blockIdx.x, tid = threadIdx.x;
  const float* s = scores + (size_t)b * NPROP;
  unsigned T = (unsigned)tg[b * 2];
  int G = tg[b * 2 + 1];
  __shared__ int pos;
  if (tid == 0) pos = 0;
  __syncthreads();
  for (int i = tid; i < NPROP; i += 256) {
    if (ordf(s[i]) > T) {
      int p = atomicAdd(&pos, 1);
      cand[(size_t)b * PRE_K + p] = i;
    }
  }
  __syncthreads();
  if (tid == 0) {
    int R = PRE_K - G, taken = 0;
    for (int i = 0; i < NPROP && taken < R; ++i)
      if (ordf(s[i]) == T) { cand[(size_t)b * PRE_K + G + taken] = i; ++taken; }
  }
}

// ---------------- per-batch greedy NMS over 2000 candidates in LDS ----------------
__global__ __launch_bounds__(256)
void nms_k(const float* __restrict__ boxes, const float* __restrict__ scores,
           const int* __restrict__ cand, float* __restrict__ rois,
           float* __restrict__ numo) {
  int b = blockIdx.x, tid = threadIdx.x;
  __shared__ float X1[PRE_K], Y1[PRE_K], X2[PRE_K], Y2[PRE_K], SC[PRE_K];
  __shared__ int VAL[PRE_K];
  __shared__ float rs[256];
  __shared__ int ri[256];
  __shared__ int npick;

  for (int i = tid; i < PRE_K; i += 256) {
    int idx = cand[(size_t)b * PRE_K + i];
    const float* bp = boxes + ((size_t)b * NPROP + idx) * 4;
    X1[i] = bp[0]; Y1[i] = bp[1]; X2[i] = bp[2]; Y2[i] = bp[3];
    float s = scores[(size_t)b * NPROP + idx];
    SC[i] = s;
    VAL[i] = (s > NEGV * 0.5f) ? 1 : 0;
  }
  if (tid == 0) npick = 0;
  __syncthreads();

  for (int t = 0; t < POST_K; ++t) {
    float bs = -3.4e38f; int bi = -1;
    for (int i = tid; i < PRE_K; i += 256)
      if (VAL[i] && SC[i] > bs) { bs = SC[i]; bi = i; }
    rs[tid] = bs; ri[tid] = bi;
    __syncthreads();
    for (int off = 128; off > 0; off >>= 1) {
      if (tid < off) {
        float s2 = rs[tid + off]; int i2 = ri[tid + off];
        bool take = (ri[tid] < 0) ||
                    (i2 >= 0 && (s2 > rs[tid] || (s2 == rs[tid] && i2 < ri[tid])));
        if (take) { rs[tid] = s2; ri[tid] = i2; }
      }
      __syncthreads();
    }
    int pick = ri[0];
    bool ok = (pick >= 0);
    if (tid == 0) {
      float* r = rois + ((size_t)b * POST_K + t) * 5;
      r[0] = (float)b;
      if (ok) { r[1] = X1[pick]; r[2] = Y1[pick]; r[3] = X2[pick]; r[4] = Y2[pick]; ++npick; }
      else    { r[1] = 0.f; r[2] = 0.f; r[3] = 0.f; r[4] = 0.f; }
    }
    if (ok) {
      float px1 = X1[pick], py1 = Y1[pick], px2 = X2[pick], py2 = Y2[pick];
      float pa = (px2 - px1 + 1.f) * (py2 - py1 + 1.f);
      for (int i = tid; i < PRE_K; i += 256) {
        if (VAL[i]) {
          float iw = fmaxf(0.f, fminf(px2, X2[i]) - fmaxf(px1, X1[i]) + 1.f);
          float ih = fmaxf(0.f, fminf(py2, Y2[i]) - fmaxf(py1, Y1[i]) + 1.f);
          float inter = iw * ih;
          float ai = (X2[i] - X1[i] + 1.f) * (Y2[i] - Y1[i] + 1.f);
          float iou = inter / (pa + ai - inter);
          if (iou > 0.7f) VAL[i] = 0;
        }
      }
    }
    __syncthreads();
  }
  if (tid == 0) numo[b] = (float)npick;
}

// ---------------- host launch ----------------
extern "C" void kernel_launch(void* const* d_in, const int* in_sizes, int n_in,
                              void* d_out, int out_size, void* d_ws, size_t ws_size,
                              hipStream_t stream) {
  (void)in_sizes; (void)n_in; (void)out_size; (void)ws_size;
  const float* x      = (const float*)d_in[0];
  const float* w1     = (const float*)d_in[1];
  const float* b1     = (const float*)d_in[2];
  const float* w_bbox = (const float*)d_in[3];
  const float* b_bbox = (const float*)d_in[4];
  const float* w_cls  = (const float*)d_in[5];
  const float* b_cls  = (const float*)d_in[6];
  float* out = (float*)d_out;

  char* ws = (char*)d_ws;
  size_t off = 0;
  auto alloc = [&](size_t bytes) -> void* {
    void* p = (void*)(ws + off);
    off += (bytes + 255) & ~(size_t)255;
    return p;
  };
  _Float16* xp  = (_Float16*)alloc((size_t)BB * PH * PW * CIN * 2);   // padded NHWC
  _Float16* w1t = (_Float16*)alloc((size_t)COUT * KTOT * 2);
  float* feat   = (float*)alloc((size_t)BB * COUT * HWo * 4);
  float* deltas = (float*)alloc((size_t)BB * 36 * HWo * 4);
  float* clsb   = (float*)alloc((size_t)BB * 18 * HWo * 4);
  float* boxes  = (float*)alloc((size_t)BB * NPROP * 4 * 4);
  float* scores = (float*)alloc((size_t)BB * NPROP * 4);
  int*   tg     = (int*)alloc((size_t)BB * 2 * 4);
  int*   cand   = (int*)alloc((size_t)BB * PRE_K * 4);

  { int n = BB * PH * PW * CIN; convert_x_k<<<(n + 255) / 256, 256, 0, stream>>>(x, xp); }
  { int n = COUT * KTOT;        convert_w_k<<<(n + 255) / 256, 256, 0, stream>>>(w1, w1t); }

  dim3 gconv((BB * HWo) / 64, COUT / 256);   // 240 x 2 blocks, 256 threads
  conv1_wmma<<<gconv, 256, 0, stream>>>(xp, w1t, b1, feat);

  { int n = BB * 36 * HWo; conv_1x1_k<<<(n + 255) / 256, 256, 0, stream>>>(feat, w_bbox, b_bbox, deltas); }
  { int n = BB * 18 * HWo; conv_cls_k<<<(n + 255) / 256, 256, 0, stream>>>(feat, w_cls, b_cls, clsb); }
  { int n = BB * NPROP;    decode_k<<<(n + 255) / 256, 256, 0, stream>>>(deltas, clsb, boxes, scores); }

  thresh_k <<<BB, 256, 0, stream>>>(scores, tg);
  compact_k<<<BB, 256, 0, stream>>>(scores, tg, cand);
  nms_k    <<<BB, 256, 0, stream>>>(boxes, scores, cand, out, out + (size_t)BB * POST_K * 5);
}
